// SelfAttention_6390911337294
// MI455X (gfx1250) — matrix operand order repaired and checked by reading, not compiled, not run on hardware
//
#include <hip/hip_runtime.h>
#include <stdint.h>

#define BATCH 32
#define CIN   512
#define COUT  256
#define NTOK  1568   // 8*14*14

static_assert(CIN  % 32 == 0, "");
static_assert(COUT % 32 == 0, "");
static_assert(NTOK % 32 == 0, "");

typedef __attribute__((ext_vector_type(16))) __bf16 v16bf;
typedef __attribute__((ext_vector_type(8)))  float  v8f;

union FragBF { v16bf v; int4 q[2]; };
union Pack8  { __bf16 h[8]; int4 q; };

// ---- WMMA fragment loaders (CDNA5 16-bit layouts, wave32) --------------------
// A operand: stored [M, K] row-major. Lane L: M = L&15; elements 0..7 -> K = k0+off+i,
// elements 8..15 -> K = k0+off+16+(i-8), off = (L&16)?8:0. Two aligned 16B loads.
__device__ __forceinline__ v16bf load_fragA(const __bf16* base, int ld, int row0, int k0, int lane) {
  FragBF f;
  const __bf16* p = base + (size_t)(row0 + (lane & 15)) * ld + (k0 + ((lane & 16) ? 8 : 0));
  f.q[0] = *(const int4*)(p);
  f.q[1] = *(const int4*)(p + 16);
  return f.v;
}
// B operand: stored [N, K] row-major ("BT"). Lane L: N = L&15; elements 0..7 -> K = k0+off+i,
// elements 8..15 -> K = k0+off+16+(i-8), off = (L&16)?8:0. Two aligned 16B loads.
__device__ __forceinline__ v16bf load_fragB(const __bf16* base, int ld, int col0, int k0, int lane) {
  FragBF f;
  const __bf16* p = base + (size_t)(col0 + (lane & 15)) * ld + (k0 + ((lane & 16) ? 8 : 0));
  f.q[0] = *(const int4*)(p);
  f.q[1] = *(const int4*)(p + 16);
  return f.v;
}

#define WMMA_BF16(a, b, c) \
  __builtin_amdgcn_wmma_f32_16x16x32_bf16(false, (a), false, (b), (short)0, (c), false, false)

// ---- D-tile stores -----------------------------------------------------------
// D lane mapping: VGPR r -> M = m0 + r + ((lane&16)?8:0), N = n0 + (lane&15).

// Store transposed [N, M] bf16 with bias: 8 consecutive elements -> one 16B store.
__device__ __forceinline__ void store_NM_bias(v8f acc, __bf16* out, int ldout, int m0, int n0,
                                              const float* bias, int lane) {
  const int n  = n0 + (lane & 15);
  const int mb = m0 + ((lane & 16) ? 8 : 0);
  Pack8 pk;
#pragma unroll
  for (int r = 0; r < 8; ++r) pk.h[r] = (__bf16)(acc[r] + bias[mb + r]);
  *(int4*)(out + (size_t)n * ldout + mb) = pk.q;
}

// Store natural [M, N] bf16 with bias (strided).
__device__ __forceinline__ void store_MN_bias(v8f acc, __bf16* out, int ld, int m0, int n0,
                                              const float* bias, int lane) {
  const int n  = n0 + (lane & 15);
  const int mb = m0 + ((lane & 16) ? 8 : 0);
#pragma unroll
  for (int r = 0; r < 8; ++r) out[(size_t)(mb + r) * ld + n] = (__bf16)(acc[r] + bias[mb + r]);
}

// Store natural [M, N] f32, scaled.
__device__ __forceinline__ void store_MN_f32(v8f acc, float* out, int ld, int m0, int n0,
                                             float scale, int lane) {
  const int n  = n0 + (lane & 15);
  const int mb = m0 + ((lane & 16) ? 8 : 0);
#pragma unroll
  for (int r = 0; r < 8; ++r) out[(size_t)(mb + r) * ld + n] = acc[r] * scale;
}

// Store natural [M, N] bf16, no bias.
__device__ __forceinline__ void store_MN(v8f acc, __bf16* out, int ld, int m0, int n0, int lane) {
  const int n  = n0 + (lane & 15);
  const int mb = m0 + ((lane & 16) ? 8 : 0);
#pragma unroll
  for (int r = 0; r < 8; ++r) out[(size_t)(mb + r) * ld + n] = (__bf16)acc[r];
}

// ---- Pre/post-processing kernels --------------------------------------------

// in: [R, C] f32 per batch -> out: [C, R] bf16 per batch (tiled via LDS).
__global__ __launch_bounds__(256) void transpose_cvt_f32_bf16(const float* __restrict__ in,
                                                              __bf16* __restrict__ out,
                                                              int R, int C) {
  __shared__ float tile[32][33];
  const int b  = blockIdx.z;
  const float* pin = in + (size_t)b * R * C;
  __bf16* pout = out + (size_t)b * R * C;
  const int c0 = blockIdx.x * 32, r0 = blockIdx.y * 32;
  for (int i = threadIdx.y; i < 32; i += 8)
    tile[i][threadIdx.x] = pin[(size_t)(r0 + i) * C + c0 + threadIdx.x];
  __syncthreads();
  for (int i = threadIdx.y; i < 32; i += 8)
    pout[(size_t)(c0 + i) * R + r0 + threadIdx.x] = (__bf16)tile[threadIdx.x][i];
}

// bf16 -> bf16 tiled transpose: in [R, C] -> out [C, R] per batch.
__global__ __launch_bounds__(256) void transpose_bf16(const __bf16* __restrict__ in,
                                                      __bf16* __restrict__ out, int R, int C) {
  __shared__ __bf16 tile[32][33];
  const int b  = blockIdx.z;
  const __bf16* pin = in + (size_t)b * R * C;
  __bf16* pout = out + (size_t)b * R * C;
  const int c0 = blockIdx.x * 32, r0 = blockIdx.y * 32;
  for (int i = threadIdx.y; i < 32; i += 8)
    tile[i][threadIdx.x] = pin[(size_t)(r0 + i) * C + c0 + threadIdx.x];
  __syncthreads();
  for (int i = threadIdx.y; i < 32; i += 8)
    pout[(size_t)(c0 + i) * R + r0 + threadIdx.x] = tile[threadIdx.x][i];
}

__global__ __launch_bounds__(256) void cvt_bf16(const float* __restrict__ in,
                                                __bf16* __restrict__ out, int n) {
  int i = blockIdx.x * 256 + threadIdx.x;
  if (i < n) out[i] = (__bf16)in[i];
}

// ---- Projections: KT/QT [N, COUT] (transposed), V [COUT, N] ------------------
__global__ __launch_bounds__(256) void proj_gemm(
    const __bf16* __restrict__ xT, const __bf16* __restrict__ qT,
    const __bf16* __restrict__ Wk, const __bf16* __restrict__ Wq, const __bf16* __restrict__ Wv,
    const float* __restrict__ bk, const float* __restrict__ bq, const float* __restrict__ bv,
    __bf16* __restrict__ KT, __bf16* __restrict__ QT, __bf16* __restrict__ V) {
  const int lane = threadIdx.x & 31;
  const int wg   = blockIdx.x * 8 + (threadIdx.x >> 5);  // 0..391
  const int oT   = wg & 7;                               // COUT/32 = 8 tiles
  const int nT   = wg >> 3;                              // NTOK/32 = 49 tiles
  const int b    = blockIdx.y;
  const int which = blockIdx.z;                          // 0:K 1:Q 2:V

  const __bf16* In = (which == 1 ? qT : xT) + (size_t)b * NTOK * CIN;  // [N, CIN]
  const __bf16* W  = (which == 0 ? Wk : (which == 1 ? Wq : Wv));       // [COUT, CIN]
  const float*  bias = (which == 0 ? bk : (which == 1 ? bq : bv));

  const int o0 = oT * 32, n0 = nT * 32;
  v8f acc[2][2] = {};
#pragma unroll 4
  for (int k0 = 0; k0 < CIN; k0 += 32) {
    v16bf a0 = load_fragA(W, CIN, o0,      k0, lane);
    v16bf a1 = load_fragA(W, CIN, o0 + 16, k0, lane);
    v16bf b0 = load_fragB(In, CIN, n0,      k0, lane);
    v16bf b1 = load_fragB(In, CIN, n0 + 16, k0, lane);
    acc[0][0] = WMMA_BF16(a0, b0, acc[0][0]);
    acc[0][1] = WMMA_BF16(a0, b1, acc[0][1]);
    acc[1][0] = WMMA_BF16(a1, b0, acc[1][0]);
    acc[1][1] = WMMA_BF16(a1, b1, acc[1][1]);
  }
  if (which == 2) {
    __bf16* out = V + (size_t)b * COUT * NTOK;  // [COUT, N]
#pragma unroll
    for (int i = 0; i < 2; ++i)
#pragma unroll
      for (int j = 0; j < 2; ++j)
        store_MN_bias(acc[i][j], out, NTOK, o0 + i * 16, n0 + j * 16, bias, lane);
  } else {
    __bf16* out = (which == 0 ? KT : QT) + (size_t)b * NTOK * COUT;  // [N, COUT]
#pragma unroll
    for (int i = 0; i < 2; ++i)
#pragma unroll
      for (int j = 0; j < 2; ++j)
        store_NM_bias(acc[i][j], out, COUT, o0 + i * 16, n0 + j * 16, bias, lane);
  }
}

// ---- scores[n,m] = (1/16) * sum_c KT[n,c] * QT[m,c] --------------------------
__global__ __launch_bounds__(256) void scores_gemm(const __bf16* __restrict__ KT,
                                                   const __bf16* __restrict__ QT,
                                                   float* __restrict__ S, int b0) {
  const int lane = threadIdx.x & 31;
  const int wg   = blockIdx.x * 8 + (threadIdx.x >> 5);
  if (wg >= 49 * 49) return;                 // whole-wave uniform exit (no barriers below)
  const int nT = wg / 49, mT = wg % 49;
  const int bl = blockIdx.y;
  const __bf16* A  = KT + (size_t)(b0 + bl) * NTOK * COUT;  // [N, COUT]
  const __bf16* Bm = QT + (size_t)(b0 + bl) * NTOK * COUT;  // [N, COUT]
  float* out = S + (size_t)bl * NTOK * NTOK;

  const int n0 = nT * 32, m0 = mT * 32;
  v8f acc[2][2] = {};
#pragma unroll
  for (int k0 = 0; k0 < COUT; k0 += 32) {
    v16bf a0 = load_fragA(A, COUT, n0,      k0, lane);
    v16bf a1 = load_fragA(A, COUT, n0 + 16, k0, lane);
    v16bf b0 = load_fragB(Bm, COUT, m0,      k0, lane);
    v16bf b1 = load_fragB(Bm, COUT, m0 + 16, k0, lane);
    acc[0][0] = WMMA_BF16(a0, b0, acc[0][0]);
    acc[0][1] = WMMA_BF16(a0, b1, acc[0][1]);
    acc[1][0] = WMMA_BF16(a1, b0, acc[1][0]);
    acc[1][1] = WMMA_BF16(a1, b1, acc[1][1]);
  }
  const float scale = 0.0625f;  // 1/sqrt(COUT)
#pragma unroll
  for (int i = 0; i < 2; ++i)
#pragma unroll
    for (int j = 0; j < 2; ++j)
      store_MN_f32(acc[i][j], out, NTOK, n0 + i * 16, m0 + j * 16, scale, lane);
}

// ---- row softmax (f32 in, bf16 out) ------------------------------------------
__global__ __launch_bounds__(256) void softmax_rows(const float* __restrict__ S,
                                                    __bf16* __restrict__ Attn) {
  __shared__ float red[256];
  const int tid = threadIdx.x;
  const float* r = S    + ((size_t)blockIdx.y * NTOK + blockIdx.x) * NTOK;
  __bf16*      o = Attn + ((size_t)blockIdx.y * NTOK + blockIdx.x) * NTOK;

  float mx = -3.4e38f;
  for (int i = tid; i < NTOK; i += 256) mx = fmaxf(mx, r[i]);
  red[tid] = mx; __syncthreads();
  for (int s = 128; s > 0; s >>= 1) { if (tid < s) red[tid] = fmaxf(red[tid], red[tid + s]); __syncthreads(); }
  mx = red[0]; __syncthreads();

  float sum = 0.f;
  for (int i = tid; i < NTOK; i += 256) sum += __expf(r[i] - mx);
  red[tid] = sum; __syncthreads();
  for (int s = 128; s > 0; s >>= 1) { if (tid < s) red[tid] += red[tid + s]; __syncthreads(); }
  const float inv = 1.0f / red[0];

  for (int i = tid; i < NTOK; i += 256) o[i] = (__bf16)(__expf(r[i] - mx) * inv);
}

// ---- O[n,c] = sum_m Attn[n,m] * V[c,m] ; O stored [N, COUT] bf16 -------------
__global__ __launch_bounds__(256) void attn_v_gemm(const __bf16* __restrict__ Attn,
                                                   const __bf16* __restrict__ V,
                                                   __bf16* __restrict__ O, int b0) {
  const int lane = threadIdx.x & 31;
  const int wg   = blockIdx.x * 8 + (threadIdx.x >> 5);  // 0..391
  const int cT   = wg & 7;                               // COUT/32
  const int nT   = wg >> 3;                              // NTOK/32
  const int bl   = blockIdx.y;
  const __bf16* A  = Attn + (size_t)bl * NTOK * NTOK;            // [N, N]
  const __bf16* Bv = V + (size_t)(b0 + bl) * COUT * NTOK;        // [COUT, N] == BT layout
  __bf16* out = O + (size_t)(b0 + bl) * NTOK * COUT;

  const int n0 = nT * 32, c0 = cT * 32;
  v8f acc[2][2] = {};
  for (int k0 = 0; k0 < NTOK; k0 += 32) {
    if (k0 + 32 < NTOK) {  // hint next tile into cache (global_prefetch_b8)
      __builtin_prefetch(A + (size_t)(n0 + (lane & 15)) * NTOK + k0 + 32, 0, 3);
      __builtin_prefetch(Bv + (size_t)(c0 + (lane & 15)) * NTOK + k0 + 32, 0, 3);
    }
    v16bf a0 = load_fragA(A, NTOK, n0,      k0, lane);
    v16bf a1 = load_fragA(A, NTOK, n0 + 16, k0, lane);
    v16bf b0 = load_fragB(Bv, NTOK, c0,      k0, lane);
    v16bf b1 = load_fragB(Bv, NTOK, c0 + 16, k0, lane);
    acc[0][0] = WMMA_BF16(a0, b0, acc[0][0]);
    acc[0][1] = WMMA_BF16(a0, b1, acc[0][1]);
    acc[1][0] = WMMA_BF16(a1, b0, acc[1][0]);
    acc[1][1] = WMMA_BF16(a1, b1, acc[1][1]);
  }
#pragma unroll
  for (int i = 0; i < 2; ++i)
#pragma unroll
    for (int j = 0; j < 2; ++j)
      store_MN(acc[i][j], out, COUT, n0 + i * 16, c0 + j * 16, lane);
}

// ---- up = Wu @ of + bu ; out = x + up * scaling ------------------------------
__global__ __launch_bounds__(256) void final_gemm(const __bf16* __restrict__ Wu,
                                                  const __bf16* __restrict__ ofT,
                                                  const float* __restrict__ bu,
                                                  const float* __restrict__ x,
                                                  const float* __restrict__ scaling,
                                                  float* __restrict__ out) {
  const int lane = threadIdx.x & 31;
  const int wg   = blockIdx.x * 8 + (threadIdx.x >> 5);  // 0..783
  const int oT   = wg & 15;                              // CIN/32 = 16
  const int nT   = wg >> 4;                              // NTOK/32 = 49
  const int b    = blockIdx.y;
  const __bf16* Bm = ofT + (size_t)b * NTOK * COUT;      // [N, COUT]
  const float s = *scaling;

  const int o0 = oT * 32, n0 = nT * 32;
  v8f acc[2][2] = {};
#pragma unroll
  for (int k0 = 0; k0 < COUT; k0 += 32) {
    v16bf a0 = load_fragA(Wu, COUT, o0,      k0, lane);
    v16bf a1 = load_fragA(Wu, COUT, o0 + 16, k0, lane);
    v16bf b0 = load_fragB(Bm, COUT, n0,      k0, lane);
    v16bf b1 = load_fragB(Bm, COUT, n0 + 16, k0, lane);
    acc[0][0] = WMMA_BF16(a0, b0, acc[0][0]);
    acc[0][1] = WMMA_BF16(a0, b1, acc[0][1]);
    acc[1][0] = WMMA_BF16(a1, b0, acc[1][0]);
    acc[1][1] = WMMA_BF16(a1, b1, acc[1][1]);
  }
  const float* xb = x + (size_t)b * CIN * NTOK;
  float* ob = out + (size_t)b * CIN * NTOK;
#pragma unroll
  for (int i = 0; i < 2; ++i)
#pragma unroll
    for (int j = 0; j < 2; ++j) {
      const int n  = n0 + j * 16 + (lane & 15);
      const int mb = o0 + i * 16 + ((lane & 16) ? 8 : 0);
#pragma unroll
      for (int r = 0; r < 8; ++r) {
        const size_t idx = (size_t)(mb + r) * NTOK + n;
        ob[idx] = xb[idx] + (acc[i][j][r] + bu[mb + r]) * s;
      }
    }
}

// ---- host orchestration ------------------------------------------------------
extern "C" void kernel_launch(void* const* d_in, const int* in_sizes, int n_in,
                              void* d_out, int out_size, void* d_ws, size_t ws_size,
                              hipStream_t stream) {
  const float* x     = (const float*)d_in[0];
  const float* query = (const float*)d_in[1];
  const float* Wk = (const float*)d_in[2];
  const float* bk = (const float*)d_in[3];
  const float* Wq = (const float*)d_in[4];
  const float* bq = (const float*)d_in[5];
  const float* Wv = (const float*)d_in[6];
  const float* bv = (const float*)d_in[7];
  const float* Wu = (const float*)d_in[8];
  const float* bu = (const float*)d_in[9];
  const float* scaling = (const float*)d_in[10];
  float* out = (float*)d_out;

  char* p = (char*)d_ws;
  auto carve = [&](size_t bytes) -> void* {
    void* r = (void*)p;
    p += (bytes + 255) & ~(size_t)255;
    return r;
  };
  const size_t eXT = (size_t)BATCH * NTOK * CIN;   // x/query transposed bf16
  const size_t eKT = (size_t)BATCH * NTOK * COUT;  // K/Q/V/O buffers
  __bf16* xT  = (__bf16*)carve(eXT * 2);
  __bf16* qT  = (__bf16*)carve(eXT * 2);
  __bf16* WkB = (__bf16*)carve((size_t)COUT * CIN * 2);
  __bf16* WqB = (__bf16*)carve((size_t)COUT * CIN * 2);
  __bf16* WvB = (__bf16*)carve((size_t)COUT * CIN * 2);
  __bf16* WuB = (__bf16*)carve((size_t)CIN * COUT * 2);
  __bf16* KT  = (__bf16*)carve(eKT * 2);
  __bf16* QT  = (__bf16*)carve(eKT * 2);
  __bf16* Vb  = (__bf16*)carve(eKT * 2);
  __bf16* Ob  = (__bf16*)carve(eKT * 2);  // raw [N, COUT] == "of" flat
  __bf16* ofT = (__bf16*)carve(eKT * 2);

  // chunk scores over batches based on remaining workspace (deterministic)
  size_t used   = (size_t)(p - (char*)d_ws);
  size_t remain = ws_size > used ? ws_size - used : 0;
  size_t perB   = (size_t)NTOK * NTOK * 6 + 1024;  // f32 scores + bf16 attn
  int chunkB = (int)(remain / perB);
  if (chunkB < 1) chunkB = 1;
  if (chunkB > BATCH) chunkB = BATCH;
  float*  Sbuf = (float*) carve((size_t)chunkB * NTOK * NTOK * 4);
  __bf16* Attn = (__bf16*)carve((size_t)chunkB * NTOK * NTOK * 2);

  dim3 tb(32, 8);
  // 1) transpose + convert inputs to bf16 [N, CIN]
  transpose_cvt_f32_bf16<<<dim3(NTOK / 32, CIN / 32, BATCH), tb, 0, stream>>>(x, xT, CIN, NTOK);
  transpose_cvt_f32_bf16<<<dim3(NTOK / 32, CIN / 32, BATCH), tb, 0, stream>>>(query, qT, CIN, NTOK);
  // 2) convert weights
  cvt_bf16<<<(COUT * CIN + 255) / 256, 256, 0, stream>>>(Wk, WkB, COUT * CIN);
  cvt_bf16<<<(COUT * CIN + 255) / 256, 256, 0, stream>>>(Wq, WqB, COUT * CIN);
  cvt_bf16<<<(COUT * CIN + 255) / 256, 256, 0, stream>>>(Wv, WvB, COUT * CIN);
  cvt_bf16<<<(CIN * COUT + 255) / 256, 256, 0, stream>>>(Wu, WuB, CIN * COUT);
  // 3) K/Q/V projections (K,Q stored transposed for the scores GEMM)
  proj_gemm<<<dim3(49, BATCH, 3), 256, 0, stream>>>(xT, qT, WkB, WqB, WvB, bk, bq, bv, KT, QT, Vb);
  // 4) attention, chunked over batches through workspace
  for (int b0 = 0; b0 < BATCH; b0 += chunkB) {
    int cb = BATCH - b0 < chunkB ? BATCH - b0 : chunkB;
    scores_gemm<<<dim3((49 * 49 + 7) / 8, cb), 256, 0, stream>>>(KT, QT, Sbuf, b0);
    softmax_rows<<<dim3(NTOK, cb), 256, 0, stream>>>(Sbuf, Attn);
    attn_v_gemm<<<dim3(49, cb), 256, 0, stream>>>(Attn, Vb, Ob, b0);
  }
  // 5) reference's raw reshape: view O flat as of[COUT, N]; transpose to ofT[N, COUT]
  transpose_bf16<<<dim3(NTOK / 32, COUT / 32, BATCH), tb, 0, stream>>>(Ob, ofT, COUT, NTOK);
  // 6) up-projection + residual
  final_gemm<<<dim3((16 * 49) / 8, BATCH), 256, 0, stream>>>(WuB, ofT, bu, x, scaling, out);
}
